// Decoder_28346784153947
// MI455X (gfx1250) — matrix operand order repaired and checked by reading, not compiled, not run on hardware
//
#include <hip/hip_runtime.h>
#include <hip/hip_bf16.h>
#include <cmath>

// ---------------------------------------------------------------------------
// Types for CDNA5 WMMA (wave32): A/B = 16 bf16 per lane, C/D = 8 f32 per lane
// ---------------------------------------------------------------------------
typedef __attribute__((ext_vector_type(16))) __bf16 bf16x16;
typedef __attribute__((ext_vector_type(8)))  float  f32x8;
typedef __attribute__((ext_vector_type(8)))  int    i32x8;

#define DEV __device__ __forceinline__

// Hardware fp32 -> bf16 convert (v_cvt_*bf16* on gfx1250)
DEV unsigned short f2bf_bits(float x) {
  return __builtin_bit_cast(unsigned short, (__bf16)x);
}

// --- packed A-fragment layout for a [32 x K] activation matrix --------------
// Fragment (mt,kt): 32 lanes x 16 bf16 contiguous per lane.
// Element (m,k): k'=k&31; lane = (m&15) + 16*hi, hi=(k'>>3)&1;
//                i = (k'&7) | ((k'&16)>>1);  offset = ((kt*2+mt)*32+lane)*16+i
DEV int apos(int m, int k) {
  int kt = k >> 5, kk = k & 31;
  int hi = (kk >> 3) & 1;
  int i  = (kk & 7) | ((kk & 16) >> 1);
  int lane = (m & 15) + (hi << 4);
  return ((kt * 2 + (m >> 4)) * 32 + lane) * 16 + i;
}
DEV float rd_packed(const unsigned short* P, int m, int k) {
  return __uint_as_float((unsigned)P[apos(m, k)] << 16);
}
DEV void wr_packed(unsigned short* P, int m, int k, float v) {
  P[apos(m, k)] = f2bf_bits(v);
}

// A fragment: one contiguous 32-byte read per lane (2x ds_load_b128)
DEV bf16x16 load_a_packed(const unsigned short* P, int mt, int kt, int lane) {
  const i32x8* p = (const i32x8*)(P + (((kt * 2 + mt) * 32 + lane) << 4));
  return __builtin_bit_cast(bf16x16, *p);
}
// B fragment: pre-packed in workspace, 32 contiguous bytes per lane (L2 hit)
DEV bf16x16 load_b_frag(const unsigned short* wpk, int KT, int kt, int nt, int lane) {
  const i32x8* p = (const i32x8*)(wpk + ((size_t)((nt * KT + kt) * 32 + lane) << 4));
  return __builtin_bit_cast(bf16x16, *p);
}
DEV f32x8 wmma_bf16(bf16x16 a, bf16x16 b, f32x8 c) {
  return __builtin_amdgcn_wmma_f32_16x16x32_bf16(false, a, false, b, (short)0, c,
                                                 false, false);
}

// ---------------------------------------------------------------------------
// Problem constants
// ---------------------------------------------------------------------------
static constexpr int B_ = 32, T_ENC = 200, ENC = 256, T_DEC = 200, DOUT = 80;

// Workspace layout (bytes)
static constexpr size_t SZ_WV   = (size_t)B_ * T_ENC * ENC * 4;  // 6,553,600
static constexpr size_t OFF_WV  = 0;
static constexpr size_t OFF_FC1 = OFF_WV  + SZ_WV;
static constexpr size_t OFF_FC2 = OFF_FC1 + 49152;
static constexpr size_t OFF_AK  = OFF_FC2 + 65536;
static constexpr size_t OFF_ARK = OFF_AK  + 589824;
static constexpr size_t OFF_U   = OFF_ARK + 393216;
static constexpr size_t OFF_W   = OFF_U   + 131072;
static constexpr size_t OFF_PRJ = OFF_W   + 131072;
static constexpr size_t OFF_R1K = OFF_PRJ + 262144;
static constexpr size_t OFF_R1R = OFF_R1K + 393216;
static constexpr size_t OFF_R2K = OFF_R1R + 393216;
static constexpr size_t OFF_R2R = OFF_R2K + 393216;
static constexpr size_t OFF_OUT = OFF_R2R + 393216;

// decoder LDS: fp32 region (GZ 24576, RH 8192, ACT 8192) + packed bf16 region
static constexpr int SM_F32   = 24576 + 8192 + 8192;                       // 40960 f
static constexpr int SM_BF16  = 8192 + 4096 + 8192 + 3 * 8192 + 3072;     // 48128 h
static constexpr size_t SMEM_BYTES = (size_t)SM_F32 * 4 + (size_t)SM_BF16 * 2; // 260,096

// ---------------------------------------------------------------------------
// Weight packer: fp32 [K x N] row-major -> bf16 B-fragment tile layout
// ---------------------------------------------------------------------------
__global__ __launch_bounds__(256) void pack_w_kernel(const float* __restrict__ src,
                                                     unsigned short* __restrict__ dst,
                                                     int Kreal, int KT, int N, int total) {
  int e = blockIdx.x * 256 + threadIdx.x;
  if (e >= total) return;
  int i = e & 15, lane = (e >> 4) & 31, tile = e >> 9;
  int kt = tile % KT, nt = tile / KT;
  int k = kt * 32 + ((lane >> 4) & 1) * 8 + (i & 7) + ((i & 8) << 1);
  int n = nt * 16 + (lane & 15);
  float v = (k < Kreal) ? src[k * N + n] : 0.0f;
  dst[e] = f2bf_bits(v);
}

// ---------------------------------------------------------------------------
// Wv = memory @ W  (6400x256 @ 256x256); one-shot, gather+convert A on the fly
// ---------------------------------------------------------------------------
__global__ __launch_bounds__(256) void wv_kernel(const float* __restrict__ mem,
                                                 const unsigned short* __restrict__ WP,
                                                 float* __restrict__ Wv) {
  int wave = blockIdx.x * 8 + (threadIdx.x >> 5);
  int lane = threadIdx.x & 31;
  if (wave >= 6400) return;
  int nt = wave & 15, mtile = wave >> 4;
  int m  = (lane & 15) + (mtile << 4);
  f32x8 acc = {};
#pragma unroll
  for (int kt = 0; kt < 8; ++kt) {
    int kb = (kt << 5) + ((lane & 16) ? 8 : 0);
    bf16x16 a;
#pragma unroll
    for (int i = 0; i < 16; ++i)
      a[i] = (__bf16)mem[(size_t)m * 256 + kb + i + ((i & 8) ? 8 : 0)];
    acc = wmma_bf16(a, load_b_frag(WP, 8, kt, nt, lane), acc);
  }
  int n  = (nt << 4) + (lane & 15);
  int mb = (mtile << 4) + ((lane & 16) ? 8 : 0);
#pragma unroll
  for (int r = 0; r < 8; ++r) Wv[(size_t)(mb + r) * 256 + n] = acc[r];
}

// ---------------------------------------------------------------------------
// Cooperative 32-wave GEMM helpers (A packed bf16, B packed bf16, C f32)
//   KT0: concat split in 32-wide fragment units (kt < KT0 -> A0, else A1)
//   OUTM bit0: write fp32 buffer, bit1: write packed bf16 buffer
// ---------------------------------------------------------------------------
template <int Kp, int KT0, int N, int RELU, int OUTM>
DEV void wg_gemm(const unsigned short* __restrict__ A0,
                 const unsigned short* __restrict__ A1,
                 const unsigned short* __restrict__ wpk,
                 const float* __restrict__ bias,
                 float* outf, int ldo, unsigned short* outp,
                 int wave, int lane) {
  constexpr int KT = Kp / 32;
  constexpr int NT = N / 16;
  for (int tile = wave; tile < 2 * NT; tile += 32) {
    int mt = tile & 1, nt = tile >> 1;
    f32x8 acc = {};
#pragma unroll
    for (int kt = 0; kt < KT; ++kt) {
      bf16x16 a = (kt < KT0) ? load_a_packed(A0, mt, kt, lane)
                             : load_a_packed(A1, mt, kt - KT0, lane);
      acc = wmma_bf16(a, load_b_frag(wpk, KT, kt, nt, lane), acc);
    }
    int n  = (nt << 4) + (lane & 15);
    int mb = (mt << 4) + ((lane & 16) ? 8 : 0);
    float bv = bias ? bias[n] : 0.0f;
#pragma unroll
    for (int r = 0; r < 8; ++r) {
      float vv = acc[r] + bv;
      if (RELU) vv = fmaxf(vv, 0.0f);
      if constexpr (OUTM & 1) outf[(mb + r) * ldo + n] = vv;
      if constexpr (OUTM & 2) wr_packed(outp, mb + r, n, vv);
    }
  }
}

// h @ rec_kernel + b1: cols [0,512) += into GZ, cols [512,768) -> RH
DEV void wg_gemm_rec(const unsigned short* __restrict__ HP,
                     const unsigned short* __restrict__ wpk,
                     const float* __restrict__ bias1,
                     float* GZ, float* RH, int wave, int lane) {
  for (int tile = wave; tile < 96; tile += 32) {
    int mt = tile & 1, nt = tile >> 1;
    f32x8 acc = {};
#pragma unroll
    for (int kt = 0; kt < 8; ++kt)
      acc = wmma_bf16(load_a_packed(HP, mt, kt, lane),
                      load_b_frag(wpk, 8, kt, nt, lane), acc);
    int n  = (nt << 4) + (lane & 15);
    int mb = (mt << 4) + ((lane & 16) ? 8 : 0);
    float bv = bias1[n];
    if (n < 512) {
#pragma unroll
      for (int r = 0; r < 8; ++r) GZ[(mb + r) * 768 + n] += acc[r] + bv;
    } else {
#pragma unroll
      for (int r = 0; r < 8; ++r) RH[(mb + r) * 256 + (n - 512)] = acc[r] + bv;
    }
  }
}

DEV void wg_gemm_out(const unsigned short* __restrict__ AP,
                     const unsigned short* __restrict__ wpk,
                     const float* __restrict__ bias,
                     float* gout, int t, int wave, int lane) {
  for (int tile = wave; tile < 10; tile += 32) {  // N=80 -> 5 n-tiles x 2 m-tiles
    int mt = tile & 1, nt = tile >> 1;
    f32x8 acc = {};
#pragma unroll
    for (int kt = 0; kt < 8; ++kt)
      acc = wmma_bf16(load_a_packed(AP, mt, kt, lane),
                      load_b_frag(wpk, 8, kt, nt, lane), acc);
    int n  = (nt << 4) + (lane & 15);
    int mb = (mt << 4) + ((lane & 16) ? 8 : 0);
    float bv = bias[n];
#pragma unroll
    for (int r = 0; r < 8; ++r)
      gout[(size_t)(mb + r) * (T_DEC * DOUT) + t * DOUT + n] = acc[r] + bv;
  }
}

// GRU elementwise (Keras reset_after): state kept packed bf16
DEV void gru_elem(const float* GZ, const float* RH, unsigned short* HP, int tid) {
  for (int i = tid; i < 8192; i += 1024) {
    int m = i >> 8, u = i & 255;
    float z  = 1.0f / (1.0f + expf(-GZ[m * 768 + u]));
    float r  = 1.0f / (1.0f + expf(-GZ[m * 768 + 256 + u]));
    float hh = tanhf(GZ[m * 768 + 512 + u] + r * RH[i]);
    float h  = rd_packed(HP, m, u);
    wr_packed(HP, m, u, z * h + (1.0f - z) * hh);
  }
}

// ---------------------------------------------------------------------------
// Persistent decoder: 1 workgroup, 32 waves, all state in LDS, 200 steps.
// ---------------------------------------------------------------------------
__global__ __launch_bounds__(1024, 1) void decoder_kernel(
    const float* __restrict__ memory, const float* __restrict__ dec_inputs,
    const float* __restrict__ Wv, const float* __restrict__ Vv,
    const float* __restrict__ fc1_b, const float* __restrict__ fc2_b,
    const float* __restrict__ attn_bias, const float* __restrict__ proj_b,
    const float* __restrict__ rnn1_bias, const float* __restrict__ rnn2_bias,
    const float* __restrict__ out_b,
    const unsigned short* __restrict__ FC1P, const unsigned short* __restrict__ FC2P,
    const unsigned short* __restrict__ AKP,  const unsigned short* __restrict__ ARKP,
    const unsigned short* __restrict__ UP,   const unsigned short* __restrict__ PRJP,
    const unsigned short* __restrict__ R1KP, const unsigned short* __restrict__ R1RP,
    const unsigned short* __restrict__ R2KP, const unsigned short* __restrict__ R2RP,
    const unsigned short* __restrict__ OUTWP, float* __restrict__ out) {
  extern __shared__ float smem[];
  float* GZ    = smem;            // 32x768 gates (24576 f)  [aliased by ALIGN]
  float* RH    = GZ + 24576;      // 32x256 rh (8192 f)      [aliased by Ua]
  float* ACT   = RH + 8192;       // 32x256 fp32 proj/r1/r2
  float* ALIGN = GZ;
  float* UA    = RH;
  unsigned short* ACTP  = (unsigned short*)(ACT + 8192);  // 32x256 packed
  unsigned short* Q2P   = ACTP + 8192;   // 32x128 packed prenet-2 out
  unsigned short* ATTNP = Q2P + 4096;    // 32x256 packed attention context (state)
  unsigned short* HAP   = ATTNP + 8192;  // 32x256 packed attention-GRU state
  unsigned short* H1P   = HAP + 8192;    // 32x256 packed decoder GRU1 state
  unsigned short* H2P   = H1P + 8192;    // 32x256 packed decoder GRU2 state
  unsigned short* XBUFP = H2P + 8192;    // 32x96  packed x_t (80 live + zero pad)

  const int tid = threadIdx.x, wave = tid >> 5, lane = tid & 31;

  for (int i = tid; i < 8192; i += 1024) { ATTNP[i] = 0; HAP[i] = 0; H1P[i] = 0; H2P[i] = 0; }
  for (int i = tid; i < 3072; i += 1024) XBUFP[i] = 0;
  __syncthreads();

  for (int t = 0; t < T_DEC; ++t) {
    // ---- load x_t into packed layout (cols >=80 stay zero) -------------
    for (int i = tid; i < B_ * DOUT; i += 1024) {
      int b = i / DOUT, j = i - b * DOUT;
      wr_packed(XBUFP, b, j, dec_inputs[((size_t)b * T_DEC + t) * DOUT + j]);
    }
    __syncthreads();
    // ---- prenet --------------------------------------------------------
    wg_gemm<96, 3, 256, 1, 2>(XBUFP, XBUFP, FC1P, fc1_b, nullptr, 0, ACTP, wave, lane);
    __syncthreads();
    wg_gemm<256, 8, 128, 1, 2>(ACTP, ACTP, FC2P, fc2_b, nullptr, 0, Q2P, wave, lane);
    __syncthreads();
    // ---- attention GRU: GZ = [q2, attn] @ attn_k + b0 ------------------
    wg_gemm<384, 4, 768, 0, 1>(Q2P, ATTNP, AKP, attn_bias, GZ, 768, nullptr, wave, lane);
    __syncthreads();
    wg_gemm_rec(HAP, ARKP, attn_bias + 768, GZ, RH, wave, lane);
    __syncthreads();
    gru_elem(GZ, RH, HAP, tid);
    __syncthreads();
    // ---- Bahdanau: Ua = h_a @ U (fp32 for the score loop) --------------
    wg_gemm<256, 8, 256, 0, 1>(HAP, HAP, UP, nullptr, UA, 256, nullptr, wave, lane);
    __syncthreads();
    // scores: tanh(Wv + Ua) . V  (Wv is L2-resident)
    for (int p = tid; p < B_ * T_ENC; p += 1024) {
      int b = p / T_ENC;
      const float* wr = Wv + (size_t)p * ENC;
      const float* ua = UA + b * ENC;
      __builtin_prefetch(wr + ENC, 0, 1);
      float s = 0.f;
      for (int d = 0; d < ENC; ++d) s += tanhf(wr[d] + ua[d]) * Vv[d];
      ALIGN[p] = s;
    }
    __syncthreads();
    // softmax over T_enc: one wave per batch row
    {
      int b = wave;
      float mx = -3.4e38f;
      for (int tt = lane; tt < T_ENC; tt += 32) mx = fmaxf(mx, ALIGN[b * T_ENC + tt]);
      for (int off = 16; off; off >>= 1) mx = fmaxf(mx, __shfl_xor(mx, off, 32));
      float s = 0.f;
      for (int tt = lane; tt < T_ENC; tt += 32) {
        float e = expf(ALIGN[b * T_ENC + tt] - mx);
        ALIGN[b * T_ENC + tt] = e;
        s += e;
      }
      for (int off = 16; off; off >>= 1) s += __shfl_xor(s, off, 32);
      float inv = 1.0f / s;
      for (int tt = lane; tt < T_ENC; tt += 32) ALIGN[b * T_ENC + tt] *= inv;
    }
    __syncthreads();
    // context: attn = align^T @ memory (coalesced over d), packed store
    {
      int b = wave;
      for (int d = lane; d < ENC; d += 32) {
        float s = 0.f;
        const float* mrow = memory + (size_t)b * T_ENC * ENC + d;
        for (int tt = 0; tt < T_ENC; ++tt) s += ALIGN[b * T_ENC + tt] * mrow[tt * ENC];
        wr_packed(ATTNP, b, d, s);
      }
    }
    __syncthreads();
    // ---- projection: [attn, h_a] @ proj_w + proj_b -> fp32 + packed ----
    wg_gemm<512, 8, 256, 0, 3>(ATTNP, HAP, PRJP, proj_b, ACT, 256, ACTP, wave, lane);
    __syncthreads();
    // ---- decoder GRU 1 + residual --------------------------------------
    wg_gemm<256, 8, 768, 0, 1>(ACTP, ACTP, R1KP, rnn1_bias, GZ, 768, nullptr, wave, lane);
    __syncthreads();
    wg_gemm_rec(H1P, R1RP, rnn1_bias + 768, GZ, RH, wave, lane);
    __syncthreads();
    gru_elem(GZ, RH, H1P, tid);
    __syncthreads();
    for (int i = tid; i < 8192; i += 1024) {
      int m = i >> 8, u = i & 255;
      float v = ACT[i] + rd_packed(H1P, m, u);
      ACT[i] = v;
      wr_packed(ACTP, m, u, v);
    }
    __syncthreads();
    // ---- decoder GRU 2 + residual --------------------------------------
    wg_gemm<256, 8, 768, 0, 1>(ACTP, ACTP, R2KP, rnn2_bias, GZ, 768, nullptr, wave, lane);
    __syncthreads();
    wg_gemm_rec(H2P, R2RP, rnn2_bias + 768, GZ, RH, wave, lane);
    __syncthreads();
    gru_elem(GZ, RH, H2P, tid);
    __syncthreads();
    for (int i = tid; i < 8192; i += 1024) {
      int m = i >> 8, u = i & 255;
      float v = ACT[i] + rd_packed(H2P, m, u);
      ACT[i] = v;
      wr_packed(ACTP, m, u, v);
    }
    __syncthreads();
    // ---- output projection --------------------------------------------
    wg_gemm_out(ACTP, OUTWP, out_b, out, t, wave, lane);
    __syncthreads();
  }
}

// ---------------------------------------------------------------------------
// Host launcher
// ---------------------------------------------------------------------------
extern "C" void kernel_launch(void* const* d_in, const int* in_sizes, int n_in,
                              void* d_out, int out_size, void* d_ws, size_t ws_size,
                              hipStream_t stream) {
  (void)in_sizes; (void)n_in; (void)out_size; (void)ws_size;
  const float* memory     = (const float*)d_in[0];
  const float* dec_inputs = (const float*)d_in[1];

  char* ws = (char*)d_ws;
  float* Wv = (float*)(ws + OFF_WV);
  unsigned short* FC1P  = (unsigned short*)(ws + OFF_FC1);
  unsigned short* FC2P  = (unsigned short*)(ws + OFF_FC2);
  unsigned short* AKP   = (unsigned short*)(ws + OFF_AK);
  unsigned short* ARKP  = (unsigned short*)(ws + OFF_ARK);
  unsigned short* UP    = (unsigned short*)(ws + OFF_U);
  unsigned short* WP    = (unsigned short*)(ws + OFF_W);
  unsigned short* PRJP  = (unsigned short*)(ws + OFF_PRJ);
  unsigned short* R1KP  = (unsigned short*)(ws + OFF_R1K);
  unsigned short* R1RP  = (unsigned short*)(ws + OFF_R1R);
  unsigned short* R2KP  = (unsigned short*)(ws + OFF_R2K);
  unsigned short* R2RP  = (unsigned short*)(ws + OFF_R2R);
  unsigned short* OUTWP = (unsigned short*)(ws + OFF_OUT);

  auto pack = [&](const void* src, unsigned short* dst, int Kreal, int KT, int N) {
    int total = KT * (N / 16) * 512;
    pack_w_kernel<<<(total + 255) / 256, 256, 0, stream>>>((const float*)src, dst,
                                                           Kreal, KT, N, total);
  };
  pack(d_in[2],  FC1P, 80,  3,  256);   // fc1_w  (K padded 80->96)
  pack(d_in[4],  FC2P, 256, 8,  128);   // fc2_w
  pack(d_in[6],  AKP,  384, 12, 768);   // attn_k
  pack(d_in[7],  ARKP, 256, 8,  768);   // attn_rk
  pack(d_in[9],  WP,   256, 8,  256);   // W
  pack(d_in[10], UP,   256, 8,  256);   // U
  pack(d_in[12], PRJP, 512, 16, 256);   // proj_w
  pack(d_in[14], R1KP, 256, 8,  768);   // rnn1_k
  pack(d_in[15], R1RP, 256, 8,  768);   // rnn1_rk
  pack(d_in[17], R2KP, 256, 8,  768);   // rnn2_k
  pack(d_in[18], R2RP, 256, 8,  768);   // rnn2_rk
  pack(d_in[20], OUTWP, 256, 8, 80);    // out_w

  // Loop-invariant Wv = memory @ W, whole-GPU WMMA
  wv_kernel<<<800, 256, 0, stream>>>(memory, WP, Wv);

  (void)hipFuncSetAttribute(reinterpret_cast<const void*>(decoder_kernel),
                            hipFuncAttributeMaxDynamicSharedMemorySize,
                            (int)SMEM_BYTES);
  decoder_kernel<<<1, 1024, SMEM_BYTES, stream>>>(
      memory, dec_inputs, Wv, (const float*)d_in[11],
      (const float*)d_in[3],  (const float*)d_in[5],  (const float*)d_in[8],
      (const float*)d_in[13], (const float*)d_in[16], (const float*)d_in[19],
      (const float*)d_in[21],
      FC1P, FC2P, AKP, ARKP, UP, PRJP, R1KP, R1RP, R2KP, R2RP, OUTWP,
      (float*)d_out);
}
